// CausalLASMonotonic_68461778698927
// MI455X (gfx1250) — matrix operand order repaired and checked by reading, not compiled, not run on hardware
//
#include <hip/hip_runtime.h>

// ---------------------------------------------------------------------------
// CDNA5 (gfx1250) bf16 WMMA implementation of the monotonic-attention LAS
// forward pass.  B=16 == WMMA M tile.  All GEMMs use v_wmma_f32_16x16x32_bf16.
// ---------------------------------------------------------------------------

typedef __attribute__((ext_vector_type(16))) __bf16 v16bf;
typedef __attribute__((ext_vector_type(8)))  float  v8f;

union BfFrag { unsigned int u[8]; v16bf v; };

__device__ __forceinline__ unsigned short f2bf(float f) {
  unsigned int u = __float_as_uint(f);
  u += 0x7FFFu + ((u >> 16) & 1u);          // round-to-nearest-even
  return (unsigned short)(u >> 16);
}
__device__ __forceinline__ unsigned int pack2bf(float a, float b) {
  return (unsigned int)f2bf(a) | ((unsigned int)f2bf(b) << 16);
}
__device__ __forceinline__ float sigmoidf(float x) { return 1.0f / (1.0f + expf(-x)); }

__device__ __forceinline__ v8f wmma_bf(const unsigned int a[8], const unsigned int b[8], v8f c) {
  BfFrag ua, ub;
#pragma unroll
  for (int i = 0; i < 8; ++i) { ua.u[i] = a[i]; ub.u[i] = b[i]; }
  return __builtin_amdgcn_wmma_f32_16x16x32_bf16(false, ua.v, false, ub.v,
                                                 (short)0, c, false, false);
}

// A fragment (16x32 bf16, ISA 7.12.2 layout) from a row-major bf16 matrix
// with leading dimension ldk (in elements).  Works for global or LDS pointers.
__device__ __forceinline__ void load_afrag(const unsigned short* A, int ldk, int mbase,
                                           int kbase, int lane, unsigned int a[8]) {
  const unsigned int* p = (const unsigned int*)A;
  int m  = mbase + (lane & 15);
  int kh = (lane >> 4) * 8;
  size_t row = ((size_t)m * (size_t)ldk) >> 1;   // ldk always even
#pragma unroll
  for (int v = 0; v < 8; ++v) {
    int k = kbase + kh + ((v < 4) ? (2 * v) : (16 + 2 * (v - 4)));
    a[v] = p[row + (k >> 1)];
  }
}

// B fragment from the pre-packed layout: [ntile][ktile][lane][8 dwords]
__device__ __forceinline__ void load_bfrag(const unsigned int* Bp, int KT, int nt, int kt,
                                           int lane, unsigned int b[8]) {
  const uint4* q = (const uint4*)(Bp + (((size_t)nt * KT + kt) << 8) + (lane << 3));
  uint4 x0 = q[0], x1 = q[1];
  b[0] = x0.x; b[1] = x0.y; b[2] = x0.z; b[3] = x0.w;
  b[4] = x1.x; b[5] = x1.y; b[6] = x1.z; b[7] = x1.w;
}

// ---------------------------------------------------------------------------
// Weight repack: W f32 [N,K] row-major (B-matrix is W^T) -> bf16 B-fragments.
// B tile 32x16: lane half h, dword v holds K pair k,k+1 with
// k = 8h + 2v (v<4) or 16 + 8h + 2(v-4) (v>=4).
// ---------------------------------------------------------------------------
__global__ void pack_b_kernel(const float* __restrict__ W, unsigned int* __restrict__ out,
                              int N, int K) {
  size_t nd = (size_t)N * K / 2;
  int KT = K >> 5;
  for (size_t i = (size_t)blockIdx.x * blockDim.x + threadIdx.x; i < nd;
       i += (size_t)gridDim.x * blockDim.x) {
    int v = (int)(i & 7);
    int lane = (int)((i >> 3) & 31);
    size_t blk = i >> 8;
    int kt = (int)(blk % KT);
    int nt = (int)(blk / KT);
    int n = nt * 16 + (lane & 15);
    int k = kt * 32 + 8 * (lane >> 4) + ((v < 4) ? (2 * v) : (16 + 2 * (v - 4)));
    size_t o = (size_t)n * K + k;
    out[i] = pack2bf(W[o], W[o + 1]);
  }
}

// causal subsample + bf16 convert: x[B=16,T=1024,F=80] -> xs[8192,160] row = t*16+b
__global__ void subsample_kernel(const float* __restrict__ x, unsigned short* __restrict__ xs) {
  for (size_t i = (size_t)blockIdx.x * blockDim.x + threadIdx.x; i < (size_t)8192 * 160;
       i += (size_t)gridDim.x * blockDim.x) {
    int fs = (int)(i % 160);
    int row = (int)(i / 160);
    int b = row & 15, t = row >> 4;
    int fr = fs / 80, ft = fs - fr * 80;
    xs[i] = f2bf(x[((size_t)b * 1024 + (size_t)(2 * t + fr)) * 80 + ft]);
  }
}

// ---------------------------------------------------------------------------
// Generic bf16 WMMA GEMM: C[M,N] = A[M,K] @ B^T + bias0 + bias1
// 256 threads / 8 waves per block; each wave computes a 16x64 strip
// (1 M-tile x 4 N-tiles).  All 4 B fragments + the A fragment are issued
// before the WMMA burst so the scheduler can use partial s_wait_loadcnt.
// Requires Nt % 4 == 0 (true for all shapes here).
// ---------------------------------------------------------------------------
__global__ __launch_bounds__(256) void gemm_bf16_kernel(
    const unsigned short* __restrict__ A, const unsigned int* __restrict__ Bp,
    const float* __restrict__ bias0, const float* __restrict__ bias1,
    float* __restrict__ Cf, unsigned short* __restrict__ Cb, int M, int N, int K) {
  int lane = threadIdx.x & 31;
  int wid = blockIdx.x * 8 + (threadIdx.x >> 5);
  int Mt = M >> 4, Nt = N >> 4, KT = K >> 5;
  int Nt4 = Nt >> 2;
  int total = Mt * Nt4;
  int nw = gridDim.x * 8;
  for (int tile = wid; tile < total; tile += nw) {
    int mt = tile / Nt4;
    int ntb = (tile - mt * Nt4) << 2;
    v8f z = {0.f, 0.f, 0.f, 0.f, 0.f, 0.f, 0.f, 0.f};
    v8f acc[4] = {z, z, z, z};
    for (int kt = 0; kt < KT; ++kt) {
      unsigned int a[8];
      unsigned int b[4][8];
      load_afrag(A, K, mt * 16, kt * 32, lane, a);
#pragma unroll
      for (int q = 0; q < 4; ++q) load_bfrag(Bp, KT, ntb + q, kt, lane, b[q]);
#pragma unroll
      for (int q = 0; q < 4; ++q) acc[q] = wmma_bf(a, b[q], acc[q]);
    }
#pragma unroll
    for (int q = 0; q < 4; ++q) {
      int col = (ntb + q) * 16 + (lane & 15);
      float bs = 0.f;
      if (bias0) bs += bias0[col];
      if (bias1) bs += bias1[col];
#pragma unroll
      for (int r = 0; r < 8; ++r) {
        int row = mt * 16 + (lane >> 4) * 8 + r;
        float v = acc[q][r] + bs;
        size_t o = (size_t)row * N + col;
        if (Cf) Cf[o] = v;
        if (Cb) Cb[o] = f2bf(v);
      }
    }
  }
}

// ---------------------------------------------------------------------------
// Persistent LSTM recurrence (one WGP, 32 waves).  gin holds x@Wih^T + b_ih +
// b_hh for the whole sequence.  Wave w owns hidden cols [16w,16w+16) of all
// four gates (tiles w, w+32, w+64, w+96) so i/f/g/o meet in-lane; c in regs,
// h in LDS as bf16 (A fragments read via ds).  B fragments are issued ahead
// of the WMMA burst; next-step gate rows are prefetched (global_prefetch).
// ---------------------------------------------------------------------------
__global__ __launch_bounds__(1024) void lstm_rec_kernel(
    const unsigned int* __restrict__ whh_pack,   // packed [128nt][16kt][256]
    const float* __restrict__ gin,               // [T2,16,2048]
    unsigned short* __restrict__ out_bf16,       // [T2*16,512]
    float* __restrict__ out_f32,                 // optional
    int T2) {
  __shared__ unsigned short hsh[16 * 512];
  int tid = threadIdx.x, lane = tid & 31, wave = tid >> 5;
  for (int i = tid; i < 16 * 512; i += 1024) hsh[i] = 0;
  float c[8];
#pragma unroll
  for (int r = 0; r < 8; ++r) c[r] = 0.f;
  int j = wave * 16 + (lane & 15);
  int kh8 = (lane >> 4) * 8;
  __syncthreads();
  for (int t = 0; t < T2; ++t) {
    // hide next step's gate-row latency behind this step's WMMAs
    if (t + 1 < T2) {
      const float* pf = gin + ((size_t)((t + 1) * 16 + kh8)) * 2048 + j;
      __builtin_prefetch(pf, 0, 1);
      __builtin_prefetch(pf + 1024, 0, 1);
    }
    v8f z = {0.f, 0.f, 0.f, 0.f, 0.f, 0.f, 0.f, 0.f};
    v8f acc[4] = {z, z, z, z};
    for (int kt = 0; kt < 16; ++kt) {
      unsigned int a[8];
      unsigned int b[4][8];
      load_afrag(hsh, 512, 0, kt * 32, lane, a);
#pragma unroll
      for (int g = 0; g < 4; ++g) load_bfrag(whh_pack, 16, wave + 32 * g, kt, lane, b[g]);
#pragma unroll
      for (int g = 0; g < 4; ++g) acc[g] = wmma_bf(a, b[g], acc[g]);
    }
    __syncthreads();   // all waves finished reading h(t-1)
#pragma unroll
    for (int r = 0; r < 8; ++r) {
      int m = kh8 + r;
      const float* gp = gin + ((size_t)(t * 16 + m)) * 2048 + j;
      float iv = acc[0][r] + gp[0];
      float fv = acc[1][r] + gp[512];
      float gv = acc[2][r] + gp[1024];
      float ov = acc[3][r] + gp[1536];
      float cc = sigmoidf(fv) * c[r] + sigmoidf(iv) * tanhf(gv);
      c[r] = cc;
      float hh = sigmoidf(ov) * tanhf(cc);
      unsigned short hb = f2bf(hh);
      hsh[m * 512 + j] = hb;
      size_t row = (size_t)(t * 16 + m);
      out_bf16[row * 512 + j] = hb;
      if (out_f32) out_f32[row * 512 + j] = hh;
    }
    __syncthreads();
  }
}

// ---------------------------------------------------------------------------
// Persistent decoder (one WGP, 32 waves, 128 steps): attention energies,
// monotonic-alpha scan, context, two WMMA LSTM cells; stages concat(h2,ctx)
// as bf16 rows for the final logits GEMM.
// ---------------------------------------------------------------------------
__global__ __launch_bounds__(1024) void decoder_kernel(
    const unsigned int* __restrict__ p_dih0, const unsigned int* __restrict__ p_dhh0,
    const unsigned int* __restrict__ p_dih1, const unsigned int* __restrict__ p_dhh1,
    const unsigned int* __restrict__ p_Wsw,
    const float* __restrict__ b_ih0, const float* __restrict__ b_hh0,
    const float* __restrict__ b_ih1, const float* __restrict__ b_hh1,
    const float* __restrict__ Wsb, const float* __restrict__ vvec,
    const float* __restrict__ enc_proj,   // [8192,128] row = tp*16+b
    const float* __restrict__ enc_f32,    // [8192,512] row = tp*16+b
    const float* __restrict__ emb,        // [8000,256]
    const int* __restrict__ y_in,         // [16,128]
    const int* __restrict__ x_lens,
    float* __restrict__ p_buf,            // [16,512]
    float* __restrict__ alphaA, float* __restrict__ alphaB,
    float* __restrict__ dp_buf,           // [16,128]
    unsigned short* __restrict__ dec_feat)// [2048,1024] row = b*128+u
{
  __shared__ unsigned short xcat[16 * 768];   // [emb(256) | context(512)] bf16
  __shared__ unsigned short h1s[16 * 512];
  __shared__ unsigned short h2s[16 * 512];
  int tid = threadIdx.x, lane = tid & 31, wave = tid >> 5;
  for (int i = tid; i < 16 * 768; i += 1024) xcat[i] = 0;
  for (int i = tid; i < 16 * 512; i += 1024) { h1s[i] = 0; h2s[i] = 0; }
  for (int i = tid; i < 8192; i += 1024) alphaA[i] = ((i & 511) == 0) ? 1.f : 0.f;
  float c1[8], c2[8];
#pragma unroll
  for (int r = 0; r < 8; ++r) { c1[r] = 0.f; c2[r] = 0.f; }
  int jcol = wave * 16 + (lane & 15);
  int kh8 = (lane >> 4) * 8;
  __syncthreads();

  for (int u = 0; u < 128; ++u) {
    const float* aprev = (u & 1) ? alphaB : alphaA;
    float* anew = (u & 1) ? alphaA : alphaB;

    // ---- Stage A: dec_proj = h2 @ Wsw^T + Wsb  (waves 0..7)
    if (wave < 8) {
      v8f acc = {0.f, 0.f, 0.f, 0.f, 0.f, 0.f, 0.f, 0.f};
      for (int kt = 0; kt < 16; ++kt) {
        unsigned int a[8], b[8];
        load_bfrag(p_Wsw, 16, wave, kt, lane, b);
        load_afrag(h2s, 512, 0, kt * 32, lane, a);
        acc = wmma_bf(a, b, acc);
      }
      int col = wave * 16 + (lane & 15);
      float bs = Wsb[col];
#pragma unroll
      for (int r = 0; r < 8; ++r) dp_buf[(kh8 + r) * 128 + col] = acc[r] + bs;
    } else if (wave == 8 && lane < 16 && u + 1 < 128) {
      // prefetch next step's embedding row while waves 0..7 do the projection
      int tok = y_in[lane * 128 + (u + 1)];
      __builtin_prefetch(emb + (size_t)tok * 256, 0, 1);
    }
    __syncthreads();

    // ---- Stage B: e = tanh(enc_proj + dec_proj) . v ; pad ; p = sigmoid(e)
    for (int i = tid; i < 8192; i += 1024) {
      int b = i & 15, tp = i >> 4;
      const float* ep = enc_proj + (size_t)i * 128;
      const float* dq = dp_buf + b * 128;
      float acc = 0.f;
      for (int a2 = 0; a2 < 128; ++a2) acc += tanhf(ep[a2] + dq[a2]) * vvec[a2];
      int el = x_lens[b] >> 1;
      if (el < 1) el = 1;
      if (el > 512) el = 512;
      if (tp >= el) acc = -3.402823466e38f;
      p_buf[b * 512 + tp] = sigmoidf(acc);
    }
    __syncthreads();

    // ---- Stage C: serial monotonic-alpha scan (16 lanes = 16 batches)
    if (tid < 16) {
      int b = tid;
      const float* pb = p_buf + b * 512;
      const float* ap = aprev + b * 512;
      float* an = anew + b * 512;
      float pj = pb[0];
      float a = pj * ap[0];
      an[0] = a;
      float rprev = (1.f - pj) / (pj + 1e-6f);
      for (int jj = 1; jj < 512; ++jj) {
        pj = pb[jj];
        a = pj * (ap[jj] + a * rprev);
        an[jj] = a;
        rprev = (1.f - pj) / (pj + 1e-6f);
      }
    }
    __syncthreads();

    // ---- Stage D: context = alpha . enc ; embedding fill
    for (int i = tid; i < 8192; i += 1024) {
      int b = i & 15, hc = i >> 4;
      const float* an = anew + b * 512;
      float acc = 0.f;
      for (int tp = 0; tp < 512; ++tp)
        acc += an[tp] * enc_f32[((size_t)tp * 16 + b) * 512 + hc];
      unsigned short cb = f2bf(acc);
      xcat[b * 768 + 256 + hc] = cb;
      dec_feat[((size_t)b * 128 + u) * 1024 + 512 + hc] = cb;
    }
    for (int i = tid; i < 4096; i += 1024) {
      int b = i >> 8, d = i & 255;
      int tok = y_in[b * 128 + u];
      xcat[b * 768 + d] = f2bf(emb[(size_t)tok * 256 + d]);
    }
    __syncthreads();

    // ---- Stage E: cell0 gates = xcat @ Wih0^T + h1 @ Whh0^T + b
    {
      v8f z = {0.f, 0.f, 0.f, 0.f, 0.f, 0.f, 0.f, 0.f};
      v8f acc[4] = {z, z, z, z};
      for (int kt = 0; kt < 24; ++kt) {
        unsigned int a[8];
        unsigned int b8[4][8];
        load_afrag(xcat, 768, 0, kt * 32, lane, a);
#pragma unroll
        for (int g = 0; g < 4; ++g) load_bfrag(p_dih0, 24, wave + 32 * g, kt, lane, b8[g]);
#pragma unroll
        for (int g = 0; g < 4; ++g) acc[g] = wmma_bf(a, b8[g], acc[g]);
      }
      for (int kt = 0; kt < 16; ++kt) {
        unsigned int a[8];
        unsigned int b8[4][8];
        load_afrag(h1s, 512, 0, kt * 32, lane, a);
#pragma unroll
        for (int g = 0; g < 4; ++g) load_bfrag(p_dhh0, 16, wave + 32 * g, kt, lane, b8[g]);
#pragma unroll
        for (int g = 0; g < 4; ++g) acc[g] = wmma_bf(a, b8[g], acc[g]);
      }
      __syncthreads();
#pragma unroll
      for (int r = 0; r < 8; ++r) {
        int m = kh8 + r;
        float iv = acc[0][r] + b_ih0[jcol] + b_hh0[jcol];
        float fv = acc[1][r] + b_ih0[jcol + 512] + b_hh0[jcol + 512];
        float gv = acc[2][r] + b_ih0[jcol + 1024] + b_hh0[jcol + 1024];
        float ov = acc[3][r] + b_ih0[jcol + 1536] + b_hh0[jcol + 1536];
        float cc = sigmoidf(fv) * c1[r] + sigmoidf(iv) * tanhf(gv);
        c1[r] = cc;
        h1s[m * 512 + jcol] = f2bf(sigmoidf(ov) * tanhf(cc));
      }
      __syncthreads();
    }

    // ---- Stage F: cell1 gates = h1 @ Wih1^T + h2 @ Whh1^T + b
    {
      v8f z = {0.f, 0.f, 0.f, 0.f, 0.f, 0.f, 0.f, 0.f};
      v8f acc[4] = {z, z, z, z};
      for (int kt = 0; kt < 16; ++kt) {
        unsigned int a[8];
        unsigned int b8[4][8];
        load_afrag(h1s, 512, 0, kt * 32, lane, a);
#pragma unroll
        for (int g = 0; g < 4; ++g) load_bfrag(p_dih1, 16, wave + 32 * g, kt, lane, b8[g]);
#pragma unroll
        for (int g = 0; g < 4; ++g) acc[g] = wmma_bf(a, b8[g], acc[g]);
      }
      for (int kt = 0; kt < 16; ++kt) {
        unsigned int a[8];
        unsigned int b8[4][8];
        load_afrag(h2s, 512, 0, kt * 32, lane, a);
#pragma unroll
        for (int g = 0; g < 4; ++g) load_bfrag(p_dhh1, 16, wave + 32 * g, kt, lane, b8[g]);
#pragma unroll
        for (int g = 0; g < 4; ++g) acc[g] = wmma_bf(a, b8[g], acc[g]);
      }
      __syncthreads();
#pragma unroll
      for (int r = 0; r < 8; ++r) {
        int m = kh8 + r;
        float iv = acc[0][r] + b_ih1[jcol] + b_hh1[jcol];
        float fv = acc[1][r] + b_ih1[jcol + 512] + b_hh1[jcol + 512];
        float gv = acc[2][r] + b_ih1[jcol + 1024] + b_hh1[jcol + 1024];
        float ov = acc[3][r] + b_ih1[jcol + 1536] + b_hh1[jcol + 1536];
        float cc = sigmoidf(fv) * c2[r] + sigmoidf(iv) * tanhf(gv);
        c2[r] = cc;
        unsigned short hb = f2bf(sigmoidf(ov) * tanhf(cc));
        h2s[m * 512 + jcol] = hb;
        dec_feat[((size_t)m * 128 + u) * 1024 + jcol] = hb;
      }
      __syncthreads();
    }
  }
}

// ---------------------------------------------------------------------------
extern "C" void kernel_launch(void* const* d_in, const int* in_sizes, int n_in,
                              void* d_out, int out_size, void* d_ws, size_t ws_size,
                              hipStream_t stream) {
  (void)in_sizes; (void)n_in; (void)out_size;
  const float* x        = (const float*)d_in[0];
  const int*   x_lens   = (const int*)d_in[1];
  const int*   y_in     = (const int*)d_in[2];
  const float* enc_in_w = (const float*)d_in[3];
  const float* enc_in_b = (const float*)d_in[4];
  const float *wih[3], *whh[3], *bih[3], *bhh[3];
  for (int l = 0; l < 3; ++l) {
    wih[l] = (const float*)d_in[5 + 4 * l + 0];
    whh[l] = (const float*)d_in[5 + 4 * l + 1];
    bih[l] = (const float*)d_in[5 + 4 * l + 2];
    bhh[l] = (const float*)d_in[5 + 4 * l + 3];
  }
  const float* W_h  = (const float*)d_in[17];
  const float* W_sw = (const float*)d_in[18];
  const float* W_sb = (const float*)d_in[19];
  const float* vvec = (const float*)d_in[20];
  const float* emb  = (const float*)d_in[21];
  const float* dih0 = (const float*)d_in[22];
  const float* dhh0 = (const float*)d_in[23];
  const float* dbi0 = (const float*)d_in[24];
  const float* dbh0 = (const float*)d_in[25];
  const float* dih1 = (const float*)d_in[26];
  const float* dhh1 = (const float*)d_in[27];
  const float* dbi1 = (const float*)d_in[28];
  const float* dbh1 = (const float*)d_in[29];
  const float* out_w = (const float*)d_in[30];
  const float* out_b = (const float*)d_in[31];

  char* base = (char*)d_ws;
  size_t off = 0;
  auto alloc = [&](size_t bytes) -> void* {
    void* p = base + off;
    off += (bytes + 255) & ~(size_t)255;
    return p;
  };
  unsigned int* p_encin = (unsigned int*)alloc((size_t)512 * 160 / 2 * 4);
  unsigned int *p_wih[3], *p_whh[3];
  for (int l = 0; l < 3; ++l) {
    p_wih[l] = (unsigned int*)alloc((size_t)2048 * 512 / 2 * 4);
    p_whh[l] = (unsigned int*)alloc((size_t)2048 * 512 / 2 * 4);
  }
  unsigned int* p_Wh   = (unsigned int*)alloc((size_t)128 * 512 / 2 * 4);
  unsigned int* p_Wsw  = (unsigned int*)alloc((size_t)128 * 512 / 2 * 4);
  unsigned int* p_dih0 = (unsigned int*)alloc((size_t)2048 * 768 / 2 * 4);
  unsigned int* p_dhh0 = (unsigned int*)alloc((size_t)2048 * 512 / 2 * 4);
  unsigned int* p_dih1 = (unsigned int*)alloc((size_t)2048 * 512 / 2 * 4);
  unsigned int* p_dhh1 = (unsigned int*)alloc((size_t)2048 * 512 / 2 * 4);
  unsigned int* p_outw = (unsigned int*)alloc((size_t)8000 * 1024 / 2 * 4);
  unsigned short* xsub = (unsigned short*)alloc((size_t)8192 * 160 * 2);
  unsigned short* seqA = (unsigned short*)alloc((size_t)8192 * 512 * 2);
  unsigned short* seqB = (unsigned short*)alloc((size_t)8192 * 512 * 2);
  float* enc_f32  = (float*)alloc((size_t)8192 * 512 * 4);
  float* gin      = (float*)alloc((size_t)8192 * 2048 * 4);
  float* enc_proj = (float*)alloc((size_t)8192 * 128 * 4);
  float* p_buf    = (float*)alloc((size_t)8192 * 4);
  float* alphaA   = (float*)alloc((size_t)8192 * 4);
  float* alphaB   = (float*)alloc((size_t)8192 * 4);
  float* dp_buf   = (float*)alloc((size_t)2048 * 4);
  unsigned short* dec_feat = (unsigned short*)alloc((size_t)2048 * 1024 * 2);
  if (off > ws_size) return;   // insufficient workspace

  auto packw = [&](const float* W, unsigned int* dst, int N, int K) {
    size_t nd = (size_t)N * K / 2;
    size_t b = (nd + 255) / 256;
    int blocks = (int)(b > 4096 ? 4096 : b);
    pack_b_kernel<<<blocks, 256, 0, stream>>>(W, dst, N, K);
  };
  packw(enc_in_w, p_encin, 512, 160);
  for (int l = 0; l < 3; ++l) {
    packw(wih[l], p_wih[l], 2048, 512);
    packw(whh[l], p_whh[l], 2048, 512);
  }
  packw(W_h, p_Wh, 128, 512);
  packw(W_sw, p_Wsw, 128, 512);
  packw(dih0, p_dih0, 2048, 768);
  packw(dhh0, p_dhh0, 2048, 512);
  packw(dih1, p_dih1, 2048, 512);
  packw(dhh1, p_dhh1, 2048, 512);
  packw(out_w, p_outw, 8000, 1024);

  subsample_kernel<<<1024, 256, 0, stream>>>(x, xsub);

  auto gemm = [&](const unsigned short* A, const unsigned int* Bp, const float* b0,
                  const float* b1, float* Cf, unsigned short* Cb, int M, int N, int K) {
    size_t strips = (size_t)(M / 16) * (N / 64);   // wave computes 16x64
    size_t b = (strips + 7) / 8;
    int blocks = (int)(b > 2048 ? 2048 : b);
    gemm_bf16_kernel<<<blocks, 256, 0, stream>>>(A, Bp, b0, b1, Cf, Cb, M, N, K);
  };

  // input projection -> seqA (bf16 only; feeds first gin GEMM)
  gemm(xsub, p_encin, enc_in_b, nullptr, nullptr, seqA, 8192, 512, 160);

  // encoder: per layer, big parallel gin GEMM then persistent recurrence
  unsigned short* s0 = seqA;
  unsigned short* s1 = seqB;
  for (int l = 0; l < 3; ++l) {
    gemm(s0, p_wih[l], bih[l], bhh[l], gin, nullptr, 8192, 2048, 512);
    lstm_rec_kernel<<<1, 1024, 0, stream>>>(p_whh[l], gin, s1,
                                            (l == 2) ? enc_f32 : nullptr, 512);
    unsigned short* t = s0; s0 = s1; s1 = t;
  }
  // s0 now holds final encoder output (bf16)
  gemm(s0, p_Wh, nullptr, nullptr, enc_proj, nullptr, 8192, 128, 512);

  decoder_kernel<<<1, 1024, 0, stream>>>(
      p_dih0, p_dhh0, p_dih1, p_dhh1, p_Wsw,
      dbi0, dbh0, dbi1, dbh1, W_sb, vvec,
      enc_proj, enc_f32, emb, y_in, x_lens,
      p_buf, alphaA, alphaB, dp_buf, dec_feat);

  // logits: [2048,1024] @ out_w^T -> d_out [B,U,V] (row = b*128+u)
  gemm(dec_feat, p_outw, out_b, nullptr, (float*)d_out, nullptr, 2048, 8000, 1024);
}
